// Encoder_38706245271727
// MI455X (gfx1250) — matrix operand order repaired and checked by reading, not compiled, not run on hardware
//
#include <hip/hip_runtime.h>

#define D 128
#define BN_EPS 1e-5f

typedef __attribute__((ext_vector_type(16))) _Float16 v16h;
typedef __attribute__((ext_vector_type(8)))  _Float16 v8h;
typedef __attribute__((ext_vector_type(8)))  float    v8f;

// ---------------- degree / normalization ----------------

__global__ void k_init_deg(float* __restrict__ deg, int N) {
    int i = blockIdx.x * blockDim.x + threadIdx.x;
    if (i < N) deg[i] = 1.0f;  // self-loop weight
}

__global__ void k_accum_deg(const int* __restrict__ col, const float* __restrict__ w,
                            float* __restrict__ deg, int E) {
    int e = blockIdx.x * blockDim.x + threadIdx.x;
    if (e < E) atomicAdd(&deg[col[e]], w[e]);
}

__global__ void k_dinv(const float* __restrict__ deg, float* __restrict__ dinv, int N) {
    int i = blockIdx.x * blockDim.x + threadIdx.x;
    if (i < N) {
        float d = deg[i];
        dinv[i] = (d > 0.0f) ? rsqrtf(d) : 0.0f;
    }
}

// ---------------- precision conversion ----------------

__global__ void k_tohalf(const float* __restrict__ x, _Float16* __restrict__ xh, int n) {
    int i = blockIdx.x * blockDim.x + threadIdx.x;
    if (i < n) xh[i] = (_Float16)x[i];
}

__global__ void k_init_agg(const float* __restrict__ bias, float* __restrict__ agg, long long total) {
    long long t = (long long)blockIdx.x * blockDim.x + threadIdx.x;
    if (t < total) agg[t] = bias[t & (D - 1)];
}

// ---------------- WMMA GEMM: h[N,128] = xh[N,128] @ Wh^T ----------------
// h[n,j] = sum_k x[n,k] * W[j,k].  B fragment for output col c, K-offset kb is
// W[c, kb..kb+15]: 16 contiguous f16 = one aligned 32-byte vector load.
// Each wave: one 16-col tile x TWO 16-row tiles (B fragment reused), 8 WMMAs.
// Block = 8 waves -> 32x128 output slab. Grid = ceil(N/32).
// Stores: single uniform branch (full slab = common case) -> unconditional.

__global__ __launch_bounds__(256) void k_gemm(const _Float16* __restrict__ xh,
                                              const _Float16* __restrict__ Wh,
                                              float* __restrict__ h, int N) {
    const int lane = threadIdx.x & 31;
    const int wave = threadIdx.x >> 5;       // 0..7 -> column tile
    const int rowBase = blockIdx.x * 32;     // two 16-row tiles
    const int c = wave * 16 + (lane & 15);
    int r0 = rowBase + (lane & 15);
    int r1 = r0 + 16;
    if (r0 > N - 1) r0 = N - 1;              // clamp: no divergence at WMMA
    if (r1 > N - 1) r1 = N - 1;
    const int aK = (lane < 16) ? 0 : 16;     // A fragment K sub-offset
    const int bK = (lane < 16) ? 0 : 16;     // B fragment K sub-offset

    const _Float16* __restrict__ arow0 = xh + (long long)r0 * D;
    const _Float16* __restrict__ arow1 = xh + (long long)r1 * D;
    const _Float16* __restrict__ brow  = Wh + (long long)c * D;

    v8f acc0 = {}, acc1 = {};

#pragma unroll
    for (int k0 = 0; k0 < D; k0 += 32) {
        // B: 16 contiguous halves of W row c  (32B aligned -> 2x b128)
        v16h b = *(const v16h*)(brow + k0 + bK);

        // A: two contiguous 8-half groups per row (16B aligned -> b128 each)
        v8h a0lo = *(const v8h*)(arow0 + k0 + aK);
        v8h a0hi = *(const v8h*)(arow0 + k0 + 8 + aK);
        v8h a1lo = *(const v8h*)(arow1 + k0 + aK);
        v8h a1hi = *(const v8h*)(arow1 + k0 + 8 + aK);
        v16h a0, a1;
#pragma unroll
        for (int i = 0; i < 8; ++i) {
            a0[i] = a0lo[i]; a0[8 + i] = a0hi[i];
            a1[i] = a1lo[i]; a1[8 + i] = a1hi[i];
        }

        acc0 = __builtin_amdgcn_wmma_f32_16x16x32_f16(false, a0, false, b,
                                                      (short)0, acc0, false, false);
        acc1 = __builtin_amdgcn_wmma_f32_16x16x32_f16(false, a1, false, b,
                                                      (short)0, acc1, false, false);
    }

    const int rOff = (lane < 16) ? 0 : 8;
    float* __restrict__ out0 = h + (long long)(rowBase + rOff) * D + c;
    float* __restrict__ out1 = h + (long long)(rowBase + 16 + rOff) * D + c;

    if (rowBase + 32 <= N) {
        // common case: whole 32-row slab in range -> unconditional stores
#pragma unroll
        for (int v = 0; v < 8; ++v) {
            out0[(long long)v * D] = acc0[v];
            out1[(long long)v * D] = acc1[v];
        }
    } else {
        // tail slab (only when N % 32 != 0)
#pragma unroll
        for (int v = 0; v < 8; ++v) {
            if (rowBase + rOff + v < N)      out0[(long long)v * D] = acc0[v];
            if (rowBase + 16 + rOff + v < N) out1[(long long)v * D] = acc1[v];
        }
    }
}

// ---------------- edge scatter (one wave32 per edge, float4 per lane) ----------------

__global__ void k_scatter(const int* __restrict__ ei, const float* __restrict__ w,
                          const float* __restrict__ dinv, const float* __restrict__ h,
                          float* __restrict__ agg, int E) {
    long long t = (long long)blockIdx.x * blockDim.x + threadIdx.x;
    int lane = (int)(t & 31);
    long long e = t >> 5;
    if (e >= E) return;
    int row = ei[e];
    int col = ei[(long long)E + e];
    float coeff = dinv[row] * w[e] * dinv[col];
    int d0 = lane * 4;
    const float4 hv = *(const float4*)(h + (long long)row * D + d0);
    float* dst = agg + (long long)col * D + d0;
    atomicAdd(dst + 0, coeff * hv.x);
    atomicAdd(dst + 1, coeff * hv.y);
    atomicAdd(dst + 2, coeff * hv.z);
    atomicAdd(dst + 3, coeff * hv.w);
}

// self loops run after edge scatter (stream-ordered) -> plain adds
__global__ void k_selfloop(const float* __restrict__ dinv, const float* __restrict__ h,
                           float* __restrict__ agg, int N) {
    long long t = (long long)blockIdx.x * blockDim.x + threadIdx.x;
    int lane = (int)(t & 31);
    long long n = t >> 5;
    if (n >= N) return;
    float di = dinv[n];
    float coeff = di * di;
    int d0 = lane * 4;
    const float4 hv = *(const float4*)(h + n * (long long)D + d0);
    float* dst = agg + n * (long long)D + d0;
    dst[0] += coeff * hv.x;
    dst[1] += coeff * hv.y;
    dst[2] += coeff * hv.z;
    dst[3] += coeff * hv.w;
}

// ---------------- batchnorm (two-pass) + relu ----------------

__global__ void k_bn_zero(float* __restrict__ bn) {
    int i = threadIdx.x;
    if (i < 2 * D) bn[i] = 0.0f;
}

__global__ void k_bn_stats(const float* __restrict__ agg, float* __restrict__ bn, int N) {
    int col = threadIdx.x & (D - 1);
    int rsub = threadIdx.x >> 7;                 // 0..1
    int rowBase = blockIdx.x * 64 + rsub * 32;
    float s = 0.0f, s2 = 0.0f;
    for (int i = 0; i < 32; ++i) {
        int r = rowBase + i;
        if (r < N) {
            float v = agg[(long long)r * D + col];
            s += v;
            s2 += v * v;
        }
    }
    atomicAdd(&bn[col], s);
    atomicAdd(&bn[D + col], s2);
}

__global__ void k_bn_apply(const float* __restrict__ agg, const float* __restrict__ bn,
                           const float* __restrict__ gamma, const float* __restrict__ beta,
                           float* __restrict__ out, int N) {
    long long t = (long long)blockIdx.x * blockDim.x + threadIdx.x;
    long long total = (long long)N * D;
    if (t >= total) return;
    int d = (int)(t & (D - 1));
    float inv_n = 1.0f / (float)N;
    float mean = bn[d] * inv_n;
    float var  = bn[D + d] * inv_n - mean * mean;       // biased variance
    float xn = (agg[t] - mean) * rsqrtf(var + BN_EPS);
    float y = xn * gamma[d] + beta[d];
    out[t] = (y > 0.0f) ? y : 0.0f;
}

// ---------------- launcher ----------------

extern "C" void kernel_launch(void* const* d_in, const int* in_sizes, int n_in,
                              void* d_out, int out_size, void* d_ws, size_t ws_size,
                              hipStream_t stream) {
    const float* x     = (const float*)d_in[0];
    const int*   ei    = (const int*)  d_in[1];
    const float* ew    = (const float*)d_in[2];
    const float* W     = (const float*)d_in[3];
    const float* bias  = (const float*)d_in[4];
    const float* gamma = (const float*)d_in[5];
    const float* beta  = (const float*)d_in[6];
    float* out = (float*)d_out;

    const int N = in_sizes[0] / D;
    const int E = in_sizes[2];
    const int L = in_sizes[3] / (D * D);

    // workspace layout (256B-aligned)
    char* ws = (char*)d_ws;
    auto alloc = [&](size_t bytes) -> char* {
        char* p = ws;
        ws += (bytes + 255) & ~(size_t)255;
        return p;
    };
    float*    deg  = (float*)   alloc((size_t)N * 4);
    float*    dinv = (float*)   alloc((size_t)N * 4);
    float*    bn   = (float*)   alloc(2 * D * 4);
    _Float16* Wh   = (_Float16*)alloc((size_t)D * D * 2);
    _Float16* xh   = (_Float16*)alloc((size_t)N * D * 2);
    float*    h    = (float*)   alloc((size_t)N * D * 4);
    float*    agg  = (float*)   alloc((size_t)N * D * 4);

    const int T = 256;
    const long long ND = (long long)N * D;

    // one-time: symmetric normalization coefficients
    k_init_deg<<<(N + T - 1) / T, T, 0, stream>>>(deg, N);
    k_accum_deg<<<(E + T - 1) / T, T, 0, stream>>>(ei + E, ew, deg, E);  // cols = ei[E..2E)
    k_dinv<<<(N + T - 1) / T, T, 0, stream>>>(deg, dinv, N);

    const float* xin = x;
    for (int layer = 0; layer < L; ++layer) {
        k_tohalf<<<(int)((ND + T - 1) / T), T, 0, stream>>>(xin, xh, (int)ND);
        k_tohalf<<<(D * D + T - 1) / T, T, 0, stream>>>(W + (size_t)layer * D * D, Wh, D * D);
        k_init_agg<<<(int)((ND + T - 1) / T), T, 0, stream>>>(bias + layer * D, agg, ND);

        k_gemm<<<(N + 31) / 32, 256, 0, stream>>>(xh, Wh, h, N);

        long long st = (long long)E * 32;
        k_scatter<<<(int)((st + T - 1) / T), T, 0, stream>>>(ei, ew, dinv, h, agg, E);
        long long sn = (long long)N * 32;
        k_selfloop<<<(int)((sn + T - 1) / T), T, 0, stream>>>(dinv, h, agg, N);

        k_bn_zero<<<1, 256, 0, stream>>>(bn);
        k_bn_stats<<<(N + 63) / 64, 256, 0, stream>>>(agg, bn, N);
        k_bn_apply<<<(int)((ND + T - 1) / T), T, 0, stream>>>(agg, bn, gamma + layer * D,
                                                              beta + layer * D, out, N);
        xin = out;  // next layer reads previous output
    }
}
